// SpatiotemporalSlotAttention_50379966382758
// MI455X (gfx1250) — matrix operand order repaired and checked by reading, not compiled, not run on hardware
//
#include <hip/hip_runtime.h>
#include <hip/hip_bf16.h>
#include <math.h>

// ---------------------------------------------------------------------------
// SpatiotemporalSlotAttention for MI455X (gfx1250, wave32, WMMA bf16)
// B=16, S=4096, D=1024, K=16 slots, 3 iterations.
// bf16 storage for all GEMM operands (k / vT each 128MB -> L2-resident),
// f32 accumulation via v_wmma_f32_16x16x32_bf16, f32 elementwise math.
// v is stored TRANSPOSED (B,D,S) so every GEMM is NT (k-contiguous on both
// operands): all fragment loads are global_load_b128.
// GEMM inner loops are software-pipelined: next k-step fragments are loaded
// before the current step's 4 independent WMMAs, so loads overlap compute.
// ---------------------------------------------------------------------------

typedef __bf16 bf16_t;
typedef __attribute__((ext_vector_type(16))) __bf16 v16bf;
typedef __attribute__((ext_vector_type(8)))  float  v8f;

#define BB  16
#define SS  4096
#define DDm 1024
#define KSL 16
#define NITERS 3
#define QK_SCALE 0.03125f   /* 1024^-0.5 */

// ---------------- WMMA helpers -------------------------------------------------

static __device__ __forceinline__ v8f wmma_bf16(v16bf a, v16bf b, v8f c) {
  // (neg_a, A, neg_b, B, c_mod, C, reuse_a, reuse_b)
  return __builtin_amdgcn_wmma_f32_16x16x32_bf16(false, a, false, b, (short)0, c,
                                                 false, false);
}

static __device__ __forceinline__ v8f zero_v8f() {
  v8f a;
#pragma unroll
  for (int i = 0; i < 8; ++i) a[i] = 0.0f;
  return a;
}

// A fragment: 16x32 bf16, rows m0..m0+15, k-contiguous source (row-major, ld).
// ISA layout: lanes 0-15 row m=lane: elems 0..7 -> K 0..7, elems 8..15 -> K 16..23
//             lanes 16-31: K offsets +8.
static __device__ __forceinline__ v16bf frag_a_bf16(const bf16_t* A, long ld,
                                                    int m0, int k0) {
  int lane = threadIdx.x & 31, r = lane & 15, h = lane >> 4;
  const bf16_t* p = A + (long)(m0 + r) * ld + k0 + h * 8;
  union { v16bf v; uint4 q[2]; } u;
  u.q[0] = *(const uint4*)(p);        // K: k0+h*8    .. +7
  u.q[1] = *(const uint4*)(p + 16);   // K: k0+16+h*8 .. +7
  return u.v;
}

// Same A fragment but converting from f32 source on the fly (tokens).
static __device__ __forceinline__ v16bf frag_a_f32(const float* A, long ld,
                                                   int m0, int k0) {
  int lane = threadIdx.x & 31, r = lane & 15, h = lane >> 4;
  const float* p = A + (long)(m0 + r) * ld + k0 + h * 8;
  float4 x0 = *(const float4*)(p);
  float4 x1 = *(const float4*)(p + 4);
  float4 x2 = *(const float4*)(p + 16);
  float4 x3 = *(const float4*)(p + 20);
  v16bf f;
  f[0]=(__bf16)x0.x; f[1]=(__bf16)x0.y; f[2]=(__bf16)x0.z; f[3]=(__bf16)x0.w;
  f[4]=(__bf16)x1.x; f[5]=(__bf16)x1.y; f[6]=(__bf16)x1.z; f[7]=(__bf16)x1.w;
  f[8]=(__bf16)x2.x; f[9]=(__bf16)x2.y; f[10]=(__bf16)x2.z; f[11]=(__bf16)x2.w;
  f[12]=(__bf16)x3.x; f[13]=(__bf16)x3.y; f[14]=(__bf16)x3.z; f[15]=(__bf16)x3.w;
  return f;
}

// B fragment for NT GEMM: B[k][n] = W[n0+n][k0+k], W row-major (N, Kd).
static __device__ __forceinline__ v16bf frag_b_nt(const bf16_t* W, long ld,
                                                  int n0, int k0) {
  int lane = threadIdx.x & 31, r = lane & 15, h = lane >> 4;
  const bf16_t* p = W + (long)(n0 + r) * ld + k0 + h * 8;
  union { v16bf v; uint4 q[2]; } u;
  u.q[0] = *(const uint4*)(p);
  u.q[1] = *(const uint4*)(p + 16);
  return u.v;
}

static __device__ __forceinline__ float gelu_exact(float x) {
  return 0.5f * x * (1.0f + erff(x * 0.70710678118654752f));
}

// One k-step worth of operand fragments (16x64 A, two 64x16 B columns).
struct FragSet { v16bf a0, a1, b00, b01, b10, b11; };

template <bool AF32>
static __device__ __forceinline__ FragSet load_frags(const bf16_t* Ab,
                                                     const float* Af,
                                                     const bf16_t* Wb, int Kd,
                                                     int m0, int n0, int k0) {
  FragSet f;
  if constexpr (AF32) {
    f.a0 = frag_a_f32(Af, Kd, m0, k0);
    f.a1 = frag_a_f32(Af, Kd, m0, k0 + 32);
  } else {
    f.a0 = frag_a_bf16(Ab, Kd, m0, k0);
    f.a1 = frag_a_bf16(Ab, Kd, m0, k0 + 32);
  }
  f.b00 = frag_b_nt(Wb, Kd, n0,      k0);
  f.b01 = frag_b_nt(Wb, Kd, n0 + 16, k0);
  f.b10 = frag_b_nt(Wb, Kd, n0,      k0 + 32);
  f.b11 = frag_b_nt(Wb, Kd, n0 + 16, k0 + 32);
  return f;
}

// ---------------- Generic NT GEMM: C = act(scale * A @ W^T + bias) ------------
// A: bf16 (M,Kd) [AF32=false] or f32 (M,Kd) [AF32=true]; W: bf16 (N,Kd).
// Outputs f32 and/or bf16. Optional batching over gridDim.z (element strides).
// TRANSC: interpret M as B*S rows and write Cb transposed per batch as
// (B, N, S): idx = ((m/S)*N + n)*S + m%S  (used to materialize vT).
// Block = 256 threads = 8 waves (2 M-waves x 4 N-waves), wave tile 16x32
// -> block tile 32x128.  Kd must be a multiple of 64.

#define ACT_NONE 0
#define ACT_GELU 1

template <bool AF32, bool TRANSC>
__global__ __launch_bounds__(256) void gemm_nt_bf16(
    const bf16_t* __restrict__ A, const float* __restrict__ A32,
    const bf16_t* __restrict__ W, const float* __restrict__ bias,
    float* __restrict__ Cf, bf16_t* __restrict__ Cb,
    int M, int N, int Kd, long strideA, long strideW, long strideC,
    float scale, int act) {
  const int bz = blockIdx.z;
  const int wave = threadIdx.x >> 5;
  const int wm = wave >> 2, wn = wave & 3;
  const int m0 = blockIdx.y * 32 + wm * 16;
  const int n0 = blockIdx.x * 128 + wn * 32;
  if (m0 >= M || n0 >= N) return;   // wave-uniform: EXEC stays all-ones

  const bf16_t* Ab = AF32 ? nullptr : A + (long)bz * strideA;
  const float*  Af = AF32 ? A32 + (long)bz * strideA : nullptr;
  const bf16_t* Wb = W + (long)bz * strideW;

  v8f acc00 = zero_v8f(), acc01 = zero_v8f();
  v8f acc10 = zero_v8f(), acc11 = zero_v8f();

  // Software pipeline: fragments for step k0 are in flight while the 4 WMMAs
  // of step k0-64 execute.
  FragSet cur = load_frags<AF32>(Ab, Af, Wb, Kd, m0, n0, 0);
  for (int k0 = 64; k0 < Kd; k0 += 64) {
    FragSet nxt = load_frags<AF32>(Ab, Af, Wb, Kd, m0, n0, k0);
    acc00 = wmma_bf16(cur.a0, cur.b00, acc00);
    acc01 = wmma_bf16(cur.a0, cur.b01, acc01);
    acc10 = wmma_bf16(cur.a1, cur.b10, acc10);
    acc11 = wmma_bf16(cur.a1, cur.b11, acc11);
    cur = nxt;
  }
  acc00 = wmma_bf16(cur.a0, cur.b00, acc00);
  acc01 = wmma_bf16(cur.a0, cur.b01, acc01);
  acc10 = wmma_bf16(cur.a1, cur.b10, acc10);
  acc11 = wmma_bf16(cur.a1, cur.b11, acc11);

#pragma unroll
  for (int i = 0; i < 8; ++i) { acc00[i] += acc10[i]; acc01[i] += acc11[i]; }

  const int lane = threadIdx.x & 31, r = lane & 15, h = lane >> 4;
#pragma unroll
  for (int t = 0; t < 2; ++t) {
    const v8f acc = t ? acc01 : acc00;
    const int n = n0 + t * 16 + r;
    const float bv = bias ? bias[n] : 0.0f;
#pragma unroll
    for (int j = 0; j < 8; ++j) {
      const int m = m0 + h * 8 + j;
      float x = acc[j] * scale + bv;
      if (act == ACT_GELU) x = gelu_exact(x);
      if constexpr (TRANSC) {
        // vT materialization: (B, N, S); consecutive j -> consecutive s.
        const long idx = ((long)(m >> 12) * N + n) * SS + (m & (SS - 1));
        Cb[idx] = (__bf16)x;
      } else {
        const long idx = (long)bz * strideC + (long)m * N + n;
        if (Cf) Cf[idx] = x;
        if (Cb) Cb[idx] = (__bf16)x;
      }
    }
  }
}

// ---------------- updates = attn_n @ v  (NT over vT, per batch) ----------------
// vT is (B, Dm, S) so both operands are s-contiguous.
// grid = (Dm/128, B), block = 128 (4 waves, each a 16x32 d-tile).
__global__ __launch_bounds__(128) void updates_gemm_vt(
    const bf16_t* __restrict__ attn_nb, const bf16_t* __restrict__ vT,
    bf16_t* __restrict__ upd) {
  const int b = blockIdx.y;
  const int wave = threadIdx.x >> 5;
  const int d0 = blockIdx.x * 128 + wave * 32;
  const bf16_t* A = attn_nb + (long)b * KSL * SS;
  const bf16_t* V = vT + (long)b * DDm * SS;

  v8f acc00 = zero_v8f(), acc01 = zero_v8f();
  v8f acc10 = zero_v8f(), acc11 = zero_v8f();

  FragSet cur = load_frags<false>(A, nullptr, V, SS, 0, d0, 0);
  for (int s0 = 64; s0 < SS; s0 += 64) {
    FragSet nxt = load_frags<false>(A, nullptr, V, SS, 0, d0, s0);
    acc00 = wmma_bf16(cur.a0, cur.b00, acc00);
    acc01 = wmma_bf16(cur.a0, cur.b01, acc01);
    acc10 = wmma_bf16(cur.a1, cur.b10, acc10);
    acc11 = wmma_bf16(cur.a1, cur.b11, acc11);
    cur = nxt;
  }
  acc00 = wmma_bf16(cur.a0, cur.b00, acc00);
  acc01 = wmma_bf16(cur.a0, cur.b01, acc01);
  acc10 = wmma_bf16(cur.a1, cur.b10, acc10);
  acc11 = wmma_bf16(cur.a1, cur.b11, acc11);

#pragma unroll
  for (int i = 0; i < 8; ++i) { acc00[i] += acc10[i]; acc01[i] += acc11[i]; }

  const int lane = threadIdx.x & 31, r = lane & 15, h = lane >> 4;
#pragma unroll
  for (int t = 0; t < 2; ++t) {
    const v8f acc = t ? acc01 : acc00;
#pragma unroll
    for (int j = 0; j < 8; ++j) {
      const int m = h * 8 + j;                 // slot index
      upd[((long)(b * KSL + m)) * DDm + d0 + t * 16 + r] = (__bf16)acc[j];
    }
  }
}

// ---------------- elementwise / reduction kernels -----------------------------

__global__ __launch_bounds__(256) void f32_to_bf16(const float* __restrict__ x,
                                                   bf16_t* __restrict__ y, long n) {
  long i = (long)blockIdx.x * 256 + threadIdx.x;
  if (i < n) y[i] = (__bf16)x[i];
}

__global__ __launch_bounds__(256) void init_slots(const float* __restrict__ mu,
                                                  float* __restrict__ slots) {
  long i = (long)blockIdx.x * 256 + threadIdx.x;   // over B*K*D
  if (i < (long)BB * KSL * DDm) slots[i] = mu[i % (KSL * DDm)];
}

// LayerNorm over rows of length D, output bf16. One block per row.
__global__ __launch_bounds__(256) void ln_rows_bf16(const float* __restrict__ X,
                                                    const float* __restrict__ g,
                                                    const float* __restrict__ bta,
                                                    bf16_t* __restrict__ Y) {
  const int row = blockIdx.x;
  const float* x = X + (long)row * DDm;
  float s = 0.0f, ss = 0.0f;
  for (int i = threadIdx.x; i < DDm; i += 256) { float v = x[i]; s += v; ss += v * v; }
  __shared__ float sh0[256], sh1[256];
  sh0[threadIdx.x] = s; sh1[threadIdx.x] = ss;
  __syncthreads();
  for (int off = 128; off > 0; off >>= 1) {
    if (threadIdx.x < off) { sh0[threadIdx.x] += sh0[threadIdx.x + off];
                             sh1[threadIdx.x] += sh1[threadIdx.x + off]; }
    __syncthreads();
  }
  const float mu  = sh0[0] * (1.0f / DDm);
  const float var = sh1[0] * (1.0f / DDm) - mu * mu;
  const float inv = rsqrtf(var + 1e-5f);
  for (int i = threadIdx.x; i < DDm; i += 256) {
    float v = (x[i] - mu) * inv * g[i] + bta[i];
    Y[(long)row * DDm + i] = (__bf16)v;
  }
}

// softmax over the 16 slots for each (b,s); writes attn (f32) into d_out region.
__global__ __launch_bounds__(256) void softmax_slots(const float* __restrict__ logits,
                                                     float* __restrict__ attn) {
  const long idx = (long)blockIdx.x * 256 + threadIdx.x;  // over B*S
  if (idx >= (long)BB * SS) return;
  const int b = (int)(idx >> 12), s = (int)(idx & (SS - 1));
  const float* p = logits + (long)b * KSL * SS + s;
  float v[KSL];
  float mx = -1e30f;
#pragma unroll
  for (int k = 0; k < KSL; ++k) { v[k] = p[(long)k * SS]; mx = fmaxf(mx, v[k]); }
  float sum = 0.0f;
#pragma unroll
  for (int k = 0; k < KSL; ++k) { v[k] = expf(v[k] - mx); sum += v[k]; }
  const float inv = 1.0f / sum;
  float* o = attn + (long)b * KSL * SS + s;
#pragma unroll
  for (int k = 0; k < KSL; ++k) o[(long)k * SS] = v[k] * inv;
}

// attn_n = attn / (rowsum over s + eps), cast to bf16. One block per (b,k) row.
__global__ __launch_bounds__(256) void attn_rownorm_bf16(const float* __restrict__ attn,
                                                         bf16_t* __restrict__ out) {
  const int row = blockIdx.x;          // 0 .. B*K-1
  const float* a = attn + (long)row * SS;
  float s = 0.0f;
  for (int i = threadIdx.x; i < SS; i += 256) s += a[i];
  __shared__ float sh[256];
  sh[threadIdx.x] = s;
  __syncthreads();
  for (int off = 128; off > 0; off >>= 1) {
    if (threadIdx.x < off) sh[threadIdx.x] += sh[threadIdx.x + off];
    __syncthreads();
  }
  const float inv = 1.0f / (sh[0] + 1e-8f);
  for (int i = threadIdx.x; i < SS; i += 256)
    out[(long)row * SS + i] = (__bf16)(a[i] * inv);
}

// Fused GRU cell (biases already folded into gi/gh by GEMM); in-place on slots.
__global__ __launch_bounds__(256) void gru_fuse(const float* __restrict__ gi,
                                                const float* __restrict__ gh,
                                                float* __restrict__ slots) {
  const long idx = (long)blockIdx.x * 256 + threadIdx.x;  // over B*K*D
  if (idx >= (long)BB * KSL * DDm) return;
  const int row = (int)(idx / DDm), d = (int)(idx % DDm);
  const long b3 = (long)row * 3 * DDm;
  const float ir = gi[b3 + d], iz = gi[b3 + DDm + d], inn = gi[b3 + 2 * DDm + d];
  const float hr = gh[b3 + d], hz = gh[b3 + DDm + d], hn  = gh[b3 + 2 * DDm + d];
  const float r = 1.0f / (1.0f + expf(-(ir + hr)));
  const float z = 1.0f / (1.0f + expf(-(iz + hz)));
  const float n = tanhf(inn + r * hn);
  const float h = slots[idx];
  slots[idx] = (1.0f - z) * n + z * h;
}

__global__ __launch_bounds__(256) void add_inplace(float* __restrict__ dst,
                                                   const float* __restrict__ src,
                                                   long n) {
  long i = (long)blockIdx.x * 256 + threadIdx.x;
  if (i < n) dst[i] += src[i];
}

// ---------------- host orchestration -----------------------------------------

static inline size_t align_up(size_t x) { return (x + 255) & ~(size_t)255; }

extern "C" void kernel_launch(void* const* d_in, const int* in_sizes, int n_in,
                              void* d_out, int out_size, void* d_ws, size_t ws_size,
                              hipStream_t stream) {
  (void)in_sizes; (void)n_in; (void)out_size; (void)ws_size;
  const float* tokens  = (const float*)d_in[0];
  const float* slot_mu = (const float*)d_in[1];
  const float* Wq = (const float*)d_in[2];  const float* bq = (const float*)d_in[3];
  const float* Wk = (const float*)d_in[4];  const float* bk = (const float*)d_in[5];
  const float* Wv = (const float*)d_in[6];  const float* bv = (const float*)d_in[7];
  const float* Wih = (const float*)d_in[8]; const float* bih = (const float*)d_in[9];
  const float* Whh = (const float*)d_in[10];const float* bhh = (const float*)d_in[11];
  const float* W1 = (const float*)d_in[12]; const float* b1 = (const float*)d_in[13];
  const float* W2 = (const float*)d_in[14]; const float* b2 = (const float*)d_in[15];
  const float* g_slots = (const float*)d_in[16]; const float* b_slots = (const float*)d_in[17];
  const float* g_mlp   = (const float*)d_in[18]; const float* b_mlp   = (const float*)d_in[19];

  float* out_slots = (float*)d_out;                        // (B,K,D)
  float* out_attn  = (float*)d_out + (long)BB * KSL * DDm; // (B,K,S)

  // -------- workspace layout (bump allocator) --------
  char* ws = (char*)d_ws;
  size_t off = 0;
  auto alloc = [&](size_t bytes) { char* p = ws + off; off += align_up(bytes); return p; };
  bf16_t* k_b   = (bf16_t*)alloc((size_t)BB * SS * DDm * 2);     // 128 MB, (B,S,D)
  bf16_t* vT_b  = (bf16_t*)alloc((size_t)BB * SS * DDm * 2);     // 128 MB, (B,D,S)
  bf16_t* Wq_b  = (bf16_t*)alloc((size_t)DDm * DDm * 2);
  bf16_t* Wk_b  = (bf16_t*)alloc((size_t)DDm * DDm * 2);
  bf16_t* Wv_b  = (bf16_t*)alloc((size_t)DDm * DDm * 2);
  bf16_t* Wih_b = (bf16_t*)alloc((size_t)3 * DDm * DDm * 2);
  bf16_t* Whh_b = (bf16_t*)alloc((size_t)3 * DDm * DDm * 2);
  bf16_t* W1_b  = (bf16_t*)alloc((size_t)4 * DDm * DDm * 2);
  bf16_t* W2_b  = (bf16_t*)alloc((size_t)4 * DDm * DDm * 2);
  float*  slots   = (float*)alloc((size_t)BB * KSL * DDm * 4);
  bf16_t* slots_b = (bf16_t*)alloc((size_t)BB * KSL * DDm * 2);
  bf16_t* lnq_b   = (bf16_t*)alloc((size_t)BB * KSL * DDm * 2);
  bf16_t* q_b     = (bf16_t*)alloc((size_t)BB * KSL * DDm * 2);
  float*  logits  = (float*)alloc((size_t)BB * KSL * SS * 4);
  bf16_t* attn_nb = (bf16_t*)alloc((size_t)BB * KSL * SS * 2);
  bf16_t* upd_b   = (bf16_t*)alloc((size_t)BB * KSL * DDm * 2);
  float*  gi      = (float*)alloc((size_t)BB * KSL * 3 * DDm * 4);
  float*  gh      = (float*)alloc((size_t)BB * KSL * 3 * DDm * 4);
  bf16_t* hmlp_b  = (bf16_t*)alloc((size_t)BB * KSL * DDm * 2);
  bf16_t* t_b     = (bf16_t*)alloc((size_t)BB * KSL * 4 * DDm * 2);
  float*  mlp_out = (float*)alloc((size_t)BB * KSL * DDm * 4);

  const long M_slots = (long)BB * KSL;         // 256
  const long nSlotEl = M_slots * DDm;          // 262144

  // -------- weight conversions (f32 -> bf16) --------
  auto conv = [&](const float* src, bf16_t* dst, long n) {
    f32_to_bf16<<<dim3((unsigned)((n + 255) / 256)), 256, 0, stream>>>(src, dst, n);
  };
  conv(Wq, Wq_b, (long)DDm * DDm);
  conv(Wk, Wk_b, (long)DDm * DDm);
  conv(Wv, Wv_b, (long)DDm * DDm);
  conv(Wih, Wih_b, (long)3 * DDm * DDm);
  conv(Whh, Whh_b, (long)3 * DDm * DDm);
  conv(W1, W1_b, (long)4 * DDm * DDm);
  conv(W2, W2_b, (long)4 * DDm * DDm);

  // -------- K/V projections: (B*S, D) x (D, D)^T, f32 tokens -> bf16 out ------
  {
    dim3 grid(DDm / 128, (BB * SS) / 32, 1);
    // k: normal (B,S,D) layout
    gemm_nt_bf16<true, false><<<grid, 256, 0, stream>>>(
        nullptr, tokens, Wk_b, bk, nullptr, k_b, BB * SS, DDm, DDm,
        0, 0, 0, 1.0f, ACT_NONE);
    // v: stored transposed as vT (B,D,S)
    gemm_nt_bf16<true, true><<<grid, 256, 0, stream>>>(
        nullptr, tokens, Wv_b, bv, nullptr, vT_b, BB * SS, DDm, DDm,
        0, 0, 0, 1.0f, ACT_NONE);
  }

  // -------- slots init --------
  init_slots<<<dim3((unsigned)((nSlotEl + 255) / 256)), 256, 0, stream>>>(slot_mu, slots);

  for (int it = 0; it < NITERS; ++it) {
    // slots_prev in bf16 (for W_hh GEMM)
    f32_to_bf16<<<dim3((unsigned)((nSlotEl + 255) / 256)), 256, 0, stream>>>(slots, slots_b, nSlotEl);
    // q = LN(slots) @ Wq^T + bq
    ln_rows_bf16<<<dim3((unsigned)M_slots), 256, 0, stream>>>(slots, g_slots, b_slots, lnq_b);
    gemm_nt_bf16<false, false><<<dim3(DDm / 128, (unsigned)(M_slots / 32), 1), 256, 0, stream>>>(
        lnq_b, nullptr, Wq_b, bq, nullptr, q_b, (int)M_slots, DDm, DDm,
        0, 0, 0, 1.0f, ACT_NONE);
    // logits[b] = (q_b @ k_b^T) * SCALE   (batched over z)
    gemm_nt_bf16<false, false><<<dim3(SS / 128, 1, BB), 256, 0, stream>>>(
        q_b, nullptr, k_b, nullptr, logits, nullptr, KSL, SS, DDm,
        (long)KSL * DDm, (long)SS * DDm, (long)KSL * SS, QK_SCALE, ACT_NONE);
    // softmax over slots -> attn (f32, final output region)
    softmax_slots<<<dim3((BB * SS) / 256), 256, 0, stream>>>(logits, out_attn);
    // attn_n (row-normalized over s) -> bf16
    attn_rownorm_bf16<<<dim3(BB * KSL), 256, 0, stream>>>(out_attn, attn_nb);
    // updates = attn_n @ v   (NT over vT, per batch)
    updates_gemm_vt<<<dim3(DDm / 128, BB), 128, 0, stream>>>(attn_nb, vT_b, upd_b);
    // GRU gates: gi = updates @ W_ih^T + b_ih ; gh = slots_prev @ W_hh^T + b_hh
    gemm_nt_bf16<false, false><<<dim3(3 * DDm / 128, (unsigned)(M_slots / 32), 1), 256, 0, stream>>>(
        upd_b, nullptr, Wih_b, bih, gi, nullptr, (int)M_slots, 3 * DDm, DDm,
        0, 0, 0, 1.0f, ACT_NONE);
    gemm_nt_bf16<false, false><<<dim3(3 * DDm / 128, (unsigned)(M_slots / 32), 1), 256, 0, stream>>>(
        slots_b, nullptr, Whh_b, bhh, gh, nullptr, (int)M_slots, 3 * DDm, DDm,
        0, 0, 0, 1.0f, ACT_NONE);
    gru_fuse<<<dim3((unsigned)((nSlotEl + 255) / 256)), 256, 0, stream>>>(gi, gh, slots);
    // MLP residual: slots += GELU(LN(slots) @ W1^T + b1) @ W2^T + b2
    ln_rows_bf16<<<dim3((unsigned)M_slots), 256, 0, stream>>>(slots, g_mlp, b_mlp, hmlp_b);
    gemm_nt_bf16<false, false><<<dim3(4 * DDm / 128, (unsigned)(M_slots / 32), 1), 256, 0, stream>>>(
        hmlp_b, nullptr, W1_b, b1, nullptr, t_b, (int)M_slots, 4 * DDm, DDm,
        0, 0, 0, 1.0f, ACT_GELU);
    gemm_nt_bf16<false, false><<<dim3(DDm / 128, (unsigned)(M_slots / 32), 1), 256, 0, stream>>>(
        t_b, nullptr, W2_b, b2, mlp_out, nullptr, (int)M_slots, DDm, 4 * DDm,
        0, 0, 0, 1.0f, ACT_NONE);
    add_inplace<<<dim3((unsigned)((nSlotEl + 255) / 256)), 256, 0, stream>>>(slots, mlp_out, nSlotEl);
  }

  // final slots -> d_out (attn already written by last softmax)
  hipMemcpyAsync(out_slots, slots, (size_t)nSlotEl * sizeof(float),
                 hipMemcpyDeviceToDevice, stream);
}
